// GAT_71451075937174
// MI455X (gfx1250) — compile-verified, matrix operand order checked by reading, not run on hardware
//
#include <hip/hip_runtime.h>
#include <math.h>

// ---------------------------------------------------------------------------
// CDNA5 WMMA types
// ---------------------------------------------------------------------------
typedef __attribute__((ext_vector_type(16))) __bf16 v16bf;
typedef __attribute__((ext_vector_type(8)))  float  v8f;

union FragBF { uint4 q[2]; v16bf bf; };

__device__ __forceinline__ unsigned f2bf_raw(float f) {
  unsigned u = __float_as_uint(f);
  u += 0x7fffu + ((u >> 16) & 1u);      // round-to-nearest-even
  return u >> 16;
}
__device__ __forceinline__ unsigned pack_bf2(float lo, float hi) {
  return f2bf_raw(lo) | (f2bf_raw(hi) << 16);
}

// monotone uint encoding of f32 (for atomicMax-based segment max)
__device__ __forceinline__ unsigned enc_f32(float f) {
  unsigned u = __float_as_uint(f);
  return (u & 0x80000000u) ? ~u : (u | 0x80000000u);
}
__device__ __forceinline__ float dec_f32(unsigned e) {
  unsigned u = (e & 0x80000000u) ? (e & 0x7fffffffu) : ~e;
  return __uint_as_float(u);
}

// ---------------------------------------------------------------------------
// bf16 WMMA GEMM: C[M,N] = A[M,K] @ B[K,N] (+ bias[N])   f32 in/out, bf16 MACs
// Block tile 128x128x32, 256 threads = 8 waves (4x2), wave tile 32x64 (2x4).
// LDS ping-pong double buffering: global loads for tile kt+1 overlap the
// WMMA work of tile kt; one barrier per k-step. Branch-free staging with
// clamped addresses + select-zero; packed 2xbf16 ds_store_b32 (K even).
// Optional split-K over gridDim.z accumulates into bias-preinitialized C
// (C buffers <=25MB -> resident in 192MB L2, atomic epilogue is cheap).
// ---------------------------------------------------------------------------
#define GBM 128
#define GBN 128
#define GBK 32
#define LDK 40   // padded row (80B = 5*16B -> b128-aligned LDS rows)

__global__ __launch_bounds__(256) void gemm_bf16_wmma(
    const float* __restrict__ A, const float* __restrict__ Bm,
    const float* __restrict__ bias, float* __restrict__ Cm,
    int M, int N, int K, int accumulate)
{
  __shared__ __align__(16) unsigned short As[2][GBM][LDK];
  __shared__ __align__(16) unsigned short Bs[2][GBN][LDK];   // transposed

  const int tid  = threadIdx.x;
  const int lane = tid & 31;
  const int wave = tid >> 5;
  const int wm   = wave >> 1;   // 0..3  (M, x32)
  const int wn   = wave & 1;    // 0..1  (N, x64)
  const int m0   = blockIdx.y * GBM;
  const int n0   = blockIdx.x * GBN;

  // split-K range (gridDim.z slices of the k-tile loop)
  const int nkAll = (K + GBK - 1) / GBK;
  const int nkPer = (nkAll + gridDim.z - 1) / gridDim.z;
  const int ktBeg = blockIdx.z * nkPer;
  const int ktEnd = min(nkAll, ktBeg + nkPer);
  if (ktBeg >= ktEnd) return;

  const int lrow  = lane & 15;
  const int khalf = (lane < 16) ? 0 : 8;   // ISA 16-bit A/B fragment layout

  // staging coordinates (pair-packed along K)
  const int ar  = tid >> 4;          // A row 0..15 (+16/iter)
  const int ac2 = (tid & 15) << 1;   // A even-K within tile
  const int bc  = tid & 127;         // B: N within tile
  const int br2 = (tid >> 7) << 1;   // B: even-K 0/2 (+4/iter)

  v8f z = {0.f,0.f,0.f,0.f,0.f,0.f,0.f,0.f};
  v8f acc[2][4] = {{z,z,z,z},{z,z,z,z}};

  float2 va[8];
  float  vb0[8], vb1[8];

  // batched register loads for one k-tile (issue all, wait once)
  auto load_regs = [&](int kt) {
    const int k0 = kt * GBK;
#pragma unroll
    for (int i = 0; i < 8; ++i) {
      int gr = m0 + ar + i * 16, gc = k0 + ac2;
      int grc = min(gr, M - 1), gcc = min(gc, K - 2);
      float2 v = *(const float2*)&A[(size_t)grc * K + gcc];
      bool inb = (gr < M) & (gc < K);
      va[i].x = inb ? v.x : 0.f;
      va[i].y = inb ? v.y : 0.f;
    }
#pragma unroll
    for (int i = 0; i < 8; ++i) {
      int rp = br2 + i * 4;
      int gr = k0 + rp, gc = n0 + bc;
      int gcc = min(gc, N - 1);
      float v0 = Bm[(size_t)min(gr,     K - 1) * N + gcc];
      float v1 = Bm[(size_t)min(gr + 1, K - 1) * N + gcc];
      vb0[i] = ((gr     < K) & (gc < N)) ? v0 : 0.f;
      vb1[i] = ((gr + 1 < K) & (gc < N)) ? v1 : 0.f;
    }
  };
  // convert + packed LDS store into buffer p
  auto store_tile = [&](int p) {
#pragma unroll
    for (int i = 0; i < 8; ++i)
      *(unsigned*)&As[p][ar + i * 16][ac2] = pack_bf2(va[i].x, va[i].y);
#pragma unroll
    for (int i = 0; i < 8; ++i)
      *(unsigned*)&Bs[p][bc][br2 + i * 4] = pack_bf2(vb0[i], vb1[i]);
  };
  // fragment loads + 8 WMMAs from buffer p
  auto compute = [&](int p) {
    FragBF fa[2], fb[4];
#pragma unroll
    for (int mi = 0; mi < 2; ++mi) {
      const unsigned short* q = &As[p][wm * 32 + mi * 16 + lrow][0];
      fa[mi].q[0] = *(const uint4*)(q + khalf);        // K = khalf..khalf+7
      fa[mi].q[1] = *(const uint4*)(q + 16 + khalf);   // K = 16+khalf..
    }
#pragma unroll
    for (int ni = 0; ni < 4; ++ni) {
      const unsigned short* q = &Bs[p][wn * 64 + ni * 16 + lrow][0];
      fb[ni].q[0] = *(const uint4*)(q + khalf);
      fb[ni].q[1] = *(const uint4*)(q + 16 + khalf);
    }
#pragma unroll
    for (int mi = 0; mi < 2; ++mi)
#pragma unroll
      for (int ni = 0; ni < 4; ++ni)
        acc[mi][ni] = __builtin_amdgcn_wmma_f32_16x16x32_bf16(
            false, fa[mi].bf, false, fb[ni].bf, (short)0, acc[mi][ni],
            false, false);
  };

  // software pipeline: prologue fills buffer 0
  load_regs(ktBeg);
  store_tile(0);
  __syncthreads();
  int p = 0;
  for (int kt = ktBeg; kt < ktEnd; ++kt) {
    if (kt + 1 < ktEnd) load_regs(kt + 1);   // global loads overlap WMMAs
    compute(p);
    if (kt + 1 < ktEnd) store_tile(p ^ 1);
    __syncthreads();
    p ^= 1;
  }

  // epilogue (C/D layout: rows r and r+8 per lane half)
#pragma unroll
  for (int mi = 0; mi < 2; ++mi)
#pragma unroll
    for (int ni = 0; ni < 4; ++ni) {
      int colN = n0 + wn * 64 + ni * 16 + (lane & 15);
      if (colN >= N) continue;
      float bv = (!accumulate && bias) ? bias[colN] : 0.f;
#pragma unroll
      for (int r = 0; r < 8; ++r) {
        int rowM = m0 + wm * 32 + mi * 16 + r + ((lane < 16) ? 0 : 8);
        if (rowM < M) {
          if (accumulate) atomicAdd(&Cm[(size_t)rowM * N + colN], acc[mi][ni][r]);
          else            Cm[(size_t)rowM * N + colN] = acc[mi][ni][r] + bv;
        }
      }
    }
}

// ---------------------------------------------------------------------------
// utility fills / reductions
// ---------------------------------------------------------------------------
__global__ void fill_f32(float* p, float v, long long n) {
  for (long long i = blockIdx.x * (long long)blockDim.x + threadIdx.x; i < n;
       i += (long long)gridDim.x * blockDim.x) p[i] = v;
}
__global__ void fill_u32(unsigned* p, unsigned v, long long n) {
  for (long long i = blockIdx.x * (long long)blockDim.x + threadIdx.x; i < n;
       i += (long long)gridDim.x * blockDim.x) p[i] = v;
}
__global__ void fill_f32_v4(float4* p, long long n4) {
  float4 z = make_float4(0.f, 0.f, 0.f, 0.f);
  for (long long i = blockIdx.x * (long long)blockDim.x + threadIdx.x; i < n4;
       i += (long long)gridDim.x * blockDim.x) p[i] = z;
}
__global__ void row_bias_init(float* x, const float* __restrict__ b, long long n, int F) {
  for (long long i = blockIdx.x * (long long)blockDim.x + threadIdx.x; i < n;
       i += (long long)gridDim.x * blockDim.x) x[i] = b[i % F];
}
__global__ void reduce_sum(const float* __restrict__ x, long long n, float* out) {
  __shared__ float sm[256];
  float s = 0.f;
  for (long long i = blockIdx.x * (long long)blockDim.x + threadIdx.x; i < n;
       i += (long long)gridDim.x * blockDim.x) s += x[i];
  sm[threadIdx.x] = s;
  __syncthreads();
  for (int w = 128; w > 0; w >>= 1) {
    if (threadIdx.x < w) sm[threadIdx.x] += sm[threadIdx.x + w];
    __syncthreads();
  }
  if (threadIdx.x == 0) atomicAdd(out, sm[0]);
}

// ---------------------------------------------------------------------------
// GATv2 edge kernels (one wave per (edge, head); H=4, C=128, F=512)
// ---------------------------------------------------------------------------
__global__ __launch_bounds__(256) void edge_logits(
    const float* __restrict__ xl, const float* __restrict__ xr,
    const float* __restrict__ We, const float* __restrict__ att,
    const int* __restrict__ src, const int* __restrict__ dst,
    const float* __restrict__ ea, const float* __restrict__ easum, float invE,
    int E, int Etot, float* __restrict__ logit, unsigned* __restrict__ segmax)
{
  int wid  = blockIdx.x * (blockDim.x >> 5) + (threadIdx.x >> 5);
  int lane = threadIdx.x & 31;
  if (wid >= Etot * 4) return;
  int e = wid >> 2, h = wid & 3;
  int s = (e < E) ? src[e] : (e - E);
  int d = (e < E) ? dst[e] : (e - E);
  float eav = (e < E) ? ea[e] : (easum[0] * invE);
  const float* pl = xl + (size_t)s * 512 + h * 128;
  const float* pr = xr + (size_t)d * 512 + h * 128;
  const float* pw = We + h * 128;
  const float* pa = att + h * 128;
  float acc = 0.f;
#pragma unroll
  for (int i = 0; i < 4; ++i) {
    int c = lane + i * 32;
    float v = pl[c] + pr[c] + eav * pw[c];
    v = (v > 0.f) ? v : 0.2f * v;          // leaky_relu(0.2)
    acc += v * pa[c];
  }
#pragma unroll
  for (int off = 16; off > 0; off >>= 1) acc += __shfl_down(acc, off, 32);
  if (lane == 0) {
    logit[wid] = acc;
    atomicMax(&segmax[d * 4 + h], enc_f32(acc));
  }
}

__global__ void edge_exp(
    const float* __restrict__ logit, const unsigned* __restrict__ segmax,
    const int* __restrict__ dst, int E, int Etot,
    float* __restrict__ ex, float* __restrict__ denom)
{
  long long t = blockIdx.x * (long long)blockDim.x + threadIdx.x;
  if (t >= (long long)Etot * 4) return;
  int e = (int)(t >> 2), h = (int)(t & 3);
  int d = (e < E) ? dst[e] : (e - E);
  float m = dec_f32(segmax[d * 4 + h]);
  float v = expf(logit[t] - m);
  ex[t] = v;
  atomicAdd(&denom[d * 4 + h], v);
}

__global__ __launch_bounds__(256) void edge_aggregate(
    const float* __restrict__ ex, const float* __restrict__ denom,
    const float* __restrict__ xl,
    const int* __restrict__ src, const int* __restrict__ dst,
    int E, int Etot, float* __restrict__ xnext, float* __restrict__ alpha)
{
  int wid  = blockIdx.x * (blockDim.x >> 5) + (threadIdx.x >> 5);
  int lane = threadIdx.x & 31;
  if (wid >= Etot * 4) return;
  int e = wid >> 2, h = wid & 3;
  int s = (e < E) ? src[e] : (e - E);
  int d = (e < E) ? dst[e] : (e - E);
  float a = ex[wid] / (denom[d * 4 + h] + 1e-16f);
  if (lane == 0) alpha[wid] = a;
  const float* pl = xl + (size_t)s * 512 + h * 128;
  float* po = xnext + (size_t)d * 512 + h * 128;
#pragma unroll
  for (int i = 0; i < 4; ++i) {
    int c = lane + i * 32;
    atomicAdd(&po[c], pl[c] * a);    // L2-resident scatter (24.6 MB << 192 MB L2)
  }
}

// ---------------------------------------------------------------------------
// GraphNorm (+ReLU) over axis 0
// ---------------------------------------------------------------------------
__global__ void col_mean(const float* __restrict__ x, int N, int F, float* mean) {
  __shared__ float sm[256];
  int c = blockIdx.x;
  float s = 0.f;
  for (int r = threadIdx.x; r < N; r += blockDim.x) s += x[(size_t)r * F + c];
  sm[threadIdx.x] = s; __syncthreads();
  for (int w = 128; w > 0; w >>= 1) {
    if (threadIdx.x < w) sm[threadIdx.x] += sm[threadIdx.x + w];
    __syncthreads();
  }
  if (threadIdx.x == 0) mean[c] = sm[0] / (float)N;
}
__global__ void col_var(const float* __restrict__ x, const float* __restrict__ mean,
                        const float* __restrict__ ms, int N, int F, float* var) {
  __shared__ float sm[256];
  int c = blockIdx.x;
  float mm = mean[c] * ms[c];
  float s = 0.f;
  for (int r = threadIdx.x; r < N; r += blockDim.x) {
    float o = x[(size_t)r * F + c] - mm;
    s += o * o;
  }
  sm[threadIdx.x] = s; __syncthreads();
  for (int w = 128; w > 0; w >>= 1) {
    if (threadIdx.x < w) sm[threadIdx.x] += sm[threadIdx.x + w];
    __syncthreads();
  }
  if (threadIdx.x == 0) var[c] = sm[0] / (float)N;
}
__global__ void norm_relu(const float* __restrict__ x, const float* __restrict__ mean,
                          const float* __restrict__ var, const float* __restrict__ ms,
                          const float* __restrict__ w, const float* __restrict__ b,
                          long long n, int F, float* __restrict__ out) {
  for (long long i = blockIdx.x * (long long)blockDim.x + threadIdx.x; i < n;
       i += (long long)gridDim.x * blockDim.x) {
    int c = (int)(i % F);
    float o = x[i] - mean[c] * ms[c];
    float y = w[c] * o * rsqrtf(var[c] + 1e-5f) + b[c];
    out[i] = (y > 0.f) ? y : 0.f;
  }
}

// ---------------------------------------------------------------------------
// attention matrix scatter + final head
// ---------------------------------------------------------------------------
__global__ void attn_scatter(const int* __restrict__ src, const int* __restrict__ dst,
                             int E, int Etot, int Nn,
                             const float* __restrict__ s1, const float* __restrict__ s2,
                             float inv, float* __restrict__ attn) {
  int e = blockIdx.x * blockDim.x + threadIdx.x;
  if (e >= Etot) return;
  int s = (e < E) ? src[e] : (e - E);
  int d = (e < E) ? dst[e] : (e - E);
  attn[(size_t)s * Nn + d] = (s1[0] + s2[0]) * inv;
}

__global__ __launch_bounds__(256) void final_head(
    const float* __restrict__ x, const int* __restrict__ idxd,
    const int* __restrict__ idxc, const float* __restrict__ W1,
    const float* __restrict__ b1, int Bn, int F, float* __restrict__ out)
{
  int wid  = blockIdx.x * (blockDim.x >> 5) + (threadIdx.x >> 5);
  int lane = threadIdx.x & 31;
  if (wid >= Bn) return;
  const float* xd = x + (size_t)idxd[wid] * F;
  const float* xc = x + (size_t)idxc[wid] * F;
  float s = 0.f;
  for (int c = lane; c < F; c += 32) s += xd[c] * W1[c] + xc[c] * W1[F + c];
#pragma unroll
  for (int off = 16; off > 0; off >>= 1) s += __shfl_down(s, off, 32);
  if (lane == 0) out[wid] = 1.f / (1.f + expf(-(s + b1[0])));
}

// ---------------------------------------------------------------------------
static inline int pick_ksplit(int M, int N, int K) {
  long long base = (long long)((N + GBN - 1) / GBN) * ((M + GBM - 1) / GBM);
  int nk = (K + GBK - 1) / GBK;
  int ks = 1;
  while (base * ks < 2048 && ks * 8 <= nk && ks < 16) ks <<= 1;
  return ks;
}

static inline void launch_gemm(const float* A, const float* B, const float* bias,
                               float* C, int M, int N, int K, hipStream_t s) {
  int ks = pick_ksplit(M, N, K);
  dim3 g((N + GBN - 1) / GBN, (M + GBM - 1) / GBM, ks);
  if (ks > 1) {
    row_bias_init<<<2048, 256, 0, s>>>(C, bias, (long long)M * N, N);
    gemm_bf16_wmma<<<g, 256, 0, s>>>(A, B, nullptr, C, M, N, K, 1);
  } else {
    gemm_bf16_wmma<<<g, 256, 0, s>>>(A, B, bias, C, M, N, K, 0);
  }
}

extern "C" void kernel_launch(void* const* d_in, const int* in_sizes, int n_in,
                              void* d_out, int out_size, void* d_ws, size_t ws_size,
                              hipStream_t stream)
{
  // ---- inputs (setup_inputs order) ----
  const float* drug  = (const float*)d_in[0];
  const float* cell  = (const float*)d_in[1];
  const float* gene  = (const float*)d_in[2];
  const int*   eidx  = (const int*)d_in[3];
  const float* eattr = (const float*)d_in[4];
  const int*   idxd  = (const int*)d_in[5];
  const int*   idxc  = (const int*)d_in[6];
  const float *Wd = (const float*)d_in[7],  *bd = (const float*)d_in[8];
  const float *Wc = (const float*)d_in[9],  *bc = (const float*)d_in[10];
  const float *Wg = (const float*)d_in[11], *bg = (const float*)d_in[12];
  const float *g1Wl = (const float*)d_in[13], *g1bl = (const float*)d_in[14];
  const float *g1Wr = (const float*)d_in[15], *g1br = (const float*)d_in[16];
  const float *g1We = (const float*)d_in[17], *g1att = (const float*)d_in[18];
  const float *g1bias = (const float*)d_in[19];
  const float *g2Wl = (const float*)d_in[20], *g2bl = (const float*)d_in[21];
  const float *g2Wr = (const float*)d_in[22], *g2br = (const float*)d_in[23];
  const float *g2We = (const float*)d_in[24], *g2att = (const float*)d_in[25];
  const float *g2bias = (const float*)d_in[26];
  const float *gn1w = (const float*)d_in[27], *gn1b = (const float*)d_in[28], *gn1ms = (const float*)d_in[29];
  const float *gn2w = (const float*)d_in[30], *gn2b = (const float*)d_in[31], *gn2ms = (const float*)d_in[32];
  const float *W1 = (const float*)d_in[33], *b1 = (const float*)d_in[34];

  const int h1 = in_sizes[8];            // 256
  const int nd = in_sizes[7] / h1;       // 500
  const int nc = in_sizes[9] / h1;       // 1000
  const int ng = in_sizes[11] / h1;      // 10500
  const int Nn = nd + nc + ng;           // 12000
  const int E  = in_sizes[4];            // 384000
  const int Bn = in_sizes[5];            // 10000
  const int F  = in_sizes[14];           // 512 = H*h2
  const int Etot = E + Nn;               // edges + self loops

  const int* srcp = eidx;
  const int* dstp = eidx + E;

  // ---- workspace layout ----
  float* f = (float*)d_ws;
  float* xP    = f; f += (size_t)Nn * h1;   // projected features [N,256]
  float* xL    = f; f += (size_t)Nn * F;    // lin_l output
  float* xR    = f; f += (size_t)Nn * F;    // lin_r output
  float* xO    = f; f += (size_t)Nn * F;    // aggregation output
  float* xN    = f; f += (size_t)Nn * F;    // normalized features
  float* logit = f; f += (size_t)Etot * 4;
  float* exb   = f; f += (size_t)Etot * 4;
  float* alph  = f; f += (size_t)Etot * 4;
  float* denom = f; f += (size_t)Nn * 4;
  float* meanb = f; f += 512;
  float* varb  = f; f += 512;
  float* scal  = f; f += 8;                 // [0]=sum(edge_attr) [1]=sum(a1) [2]=sum(a2)
  unsigned* segm = (unsigned*)f;            // [Nn*4]

  const long long tasks = (long long)Etot * 4;
  const int wblocks = (int)((tasks + 7) / 8);          // 8 waves / block
  const int tblocks = (int)((tasks + 255) / 256);
  const float invE  = 1.0f / (float)E;
  const float invA  = 1.0f / (float)tasks;             // mean over [Etot, H]

  fill_f32<<<1, 64, 0, stream>>>(scal, 0.f, 8);
  reduce_sum<<<512, 256, 0, stream>>>(eattr, E, &scal[0]);

  // ---- input projections -> xP [N, h1] ----
  launch_gemm(drug, Wd, bd, xP,                          nd, h1, nd, stream);
  launch_gemm(cell, Wc, bc, xP + (size_t)nd * h1,        nc, h1, nc, stream);
  launch_gemm(gene, Wg, bg, xP + (size_t)(nd + nc) * h1, ng, h1, ng, stream);

  // ================= GATv2 layer 1 =================
  launch_gemm(xP, g1Wl, g1bl, xL, Nn, F, h1, stream);
  launch_gemm(xP, g1Wr, g1br, xR, Nn, F, h1, stream);
  fill_u32<<<256, 256, 0, stream>>>(segm, 0u, (long long)Nn * 4);
  fill_f32<<<256, 256, 0, stream>>>(denom, 0.f, (long long)Nn * 4);
  edge_logits<<<wblocks, 256, 0, stream>>>(xL, xR, g1We, g1att, srcp, dstp,
                                           eattr, &scal[0], invE, E, Etot, logit, segm);
  edge_exp<<<tblocks, 256, 0, stream>>>(logit, segm, dstp, E, Etot, exb, denom);
  row_bias_init<<<4096, 256, 0, stream>>>(xO, g1bias, (long long)Nn * F, F);
  edge_aggregate<<<wblocks, 256, 0, stream>>>(exb, denom, xL, srcp, dstp, E, Etot, xO, alph);
  reduce_sum<<<512, 256, 0, stream>>>(alph, tasks, &scal[1]);
  col_mean<<<F, 256, 0, stream>>>(xO, Nn, F, meanb);
  col_var<<<F, 256, 0, stream>>>(xO, meanb, gn1ms, Nn, F, varb);
  norm_relu<<<4096, 256, 0, stream>>>(xO, meanb, varb, gn1ms, gn1w, gn1b,
                                      (long long)Nn * F, F, xN);

  // ================= GATv2 layer 2 =================
  launch_gemm(xN, g2Wl, g2bl, xL, Nn, F, F, stream);
  launch_gemm(xN, g2Wr, g2br, xR, Nn, F, F, stream);
  fill_u32<<<256, 256, 0, stream>>>(segm, 0u, (long long)Nn * 4);
  fill_f32<<<256, 256, 0, stream>>>(denom, 0.f, (long long)Nn * 4);
  edge_logits<<<wblocks, 256, 0, stream>>>(xL, xR, g2We, g2att, srcp, dstp,
                                           eattr, &scal[0], invE, E, Etot, logit, segm);
  edge_exp<<<tblocks, 256, 0, stream>>>(logit, segm, dstp, E, Etot, exb, denom);
  row_bias_init<<<4096, 256, 0, stream>>>(xO, g2bias, (long long)Nn * F, F);
  edge_aggregate<<<wblocks, 256, 0, stream>>>(exb, denom, xL, srcp, dstp, E, Etot, xO, alph);
  reduce_sum<<<512, 256, 0, stream>>>(alph, tasks, &scal[2]);
  col_mean<<<F, 256, 0, stream>>>(xO, Nn, F, meanb);
  col_var<<<F, 256, 0, stream>>>(xO, meanb, gn2ms, Nn, F, varb);
  norm_relu<<<4096, 256, 0, stream>>>(xO, meanb, varb, gn2ms, gn2w, gn2b,
                                      (long long)Nn * F, F, xN);

  // ================= outputs =================
  float* outp = (float*)d_out;                 // [B] sigmoid scores
  float* attn = outp + Bn;                     // [N*N] attention matrix
  final_head<<<(Bn + 7) / 8, 256, 0, stream>>>(xN, idxd, idxc, W1, b1, Bn, F, outp);
  fill_f32_v4<<<8192, 256, 0, stream>>>((float4*)attn, (long long)Nn * Nn / 4);
  attn_scatter<<<(Etot + 255) / 256, 256, 0, stream>>>(srcp, dstp, E, Etot, Nn,
                                                       &scal[1], &scal[2], invA, attn);
}
